// InvariantPointAttention_4183298146715
// MI455X (gfx1250) — matrix operand order repaired and checked by reading, not compiled, not run on hardware
//
#include <hip/hip_runtime.h>
#include <hip/hip_bf16.h>
#include <math.h>
#include <stdint.h>

// ---------------- types / constants ----------------
typedef __attribute__((ext_vector_type(16))) __bf16 v16bf;
typedef __attribute__((ext_vector_type(8)))  __bf16 v8bf;
typedef __attribute__((ext_vector_type(8)))  float  v8f;
typedef __attribute__((ext_vector_type(4)))  float  f4;

#define NRES   768
#define HEADS  12
#define CH     16
#define CS     384
#define CZ     128
#define NP     4
#define NPV    8
#define DCAT   2112      // 192 + 288 + 96 + 1536
#define OFF_V    0
#define OFF_VP   192
#define OFF_NORM 480
#define OFF_PAIR 576

__device__ __forceinline__ v8f wmma_bf16(v16bf a, v16bf b, v8f c) {
  // D = A(16x32 bf16) * B(32x16 bf16) + C(16x16 f32)
  return __builtin_amdgcn_wmma_f32_16x16x32_bf16(
      false, a, false, b, (short)0, c, false, false);
}

// gfx1250 async copy: global -> LDS, 16B per lane, tracked by ASYNCcnt
__device__ __forceinline__ void async_load_b128_to_lds(unsigned lds_byte, const float* g) {
  asm volatile("global_load_async_to_lds_b128 %0, %1, off"
               :: "v"(lds_byte), "v"((unsigned long long)(uintptr_t)g)
               : "memory");
}
__device__ __forceinline__ void wait_async0() {
  asm volatile("s_wait_asynccnt 0x0" ::: "memory");
}

// ---------------- fragment builders (contiguous k = 16h + s, same order for A and B) ----------------
// A (16x32 MxK): lane<16 -> row=lane, K=0..15; lane>=16 -> row=lane-16, K=16..31
// Per lane the 16 elements are two contiguous runs of 8 -> vectorized loads.
__device__ __forceinline__ v16bf frag_a_f32_vec(const float* __restrict__ A, long rowStride,
                                                int row, int khalf, int k0) {
  const float* p = A + (long)row * rowStride + k0 + 2 * khalf;
  f4 a0 = *(const f4*)(p);
  f4 a1 = *(const f4*)(p + 4);
  f4 a2 = *(const f4*)(p + 8);
  f4 a3 = *(const f4*)(p + 12);
  v16bf f;
#pragma unroll
  for (int e = 0; e < 4; ++e) {
    f[e]      = (__bf16)a0[e];
    f[e + 4]  = (__bf16)a1[e];
    f[e + 8]  = (__bf16)a2[e];
    f[e + 12] = (__bf16)a3[e];
  }
  return f;
}
__device__ __forceinline__ v16bf frag_a_bf16_vec(const __bf16* __restrict__ A, long rowStride,
                                                 int row, int khalf, int k0) {
  const __bf16* p = A + (long)row * rowStride + k0 + 2 * khalf;
  v8bf a0 = *(const v8bf*)(p);
  v8bf a1 = *(const v8bf*)(p + 8);
  v16bf f;
#pragma unroll
  for (int e = 0; e < 8; ++e) { f[e] = a0[e]; f[e + 8] = a1[e]; }
  return f;
}
// B (32x16 KxN): lane<16 -> col=lane, K sequential 0..15; lane>=16 -> col=lane-16, K=16..31.
// From a transposed bf16 matrix WT[n][K] each lane reads 16 contiguous bf16.
__device__ __forceinline__ v16bf frag_b_bf16T(const __bf16* __restrict__ WT, long K,
                                              int col, int kbase, int k0) {
  const __bf16* p = WT + (long)col * K + k0 + kbase;
  v8bf b0 = *(const v8bf*)(p);
  v8bf b1 = *(const v8bf*)(p + 8);
  v16bf f;
#pragma unroll
  for (int e = 0; e < 8; ++e) { f[e] = b0[e]; f[e + 8] = b1[e]; }
  return f;
}

// ---------------- transpose weights to bf16: WT[n*K + k] = W[k*N + n] ----------------
__global__ void transpose_bf16_kernel(const float* __restrict__ W, int K, int N,
                                      __bf16* __restrict__ WT) {
  long idx = (long)blockIdx.x * 256 + threadIdx.x;
  if (idx < (long)K * N) {
    long k = idx / N, n = idx % N;
    WT[n * (long)K + k] = (__bf16)W[idx];
  }
}

// ---------------- generic GEMM + bias: out = A(f32) @ W + b, W pre-transposed bf16 --------
// optional bf16 transposed copy of the output (outT[n*M + m]) for later B-side use
__global__ void gemm_bias_kernel(const float* __restrict__ A, int lda,
                                 const __bf16* __restrict__ WT,
                                 const float* __restrict__ bias,
                                 float* __restrict__ out, int ldo, int K,
                                 __bf16* __restrict__ outT, int M) {
  const int mt = blockIdx.x, nt = blockIdx.y;
  const int lane  = threadIdx.x & 31;
  const int row   = lane & 15;
  const int khalf = (lane >> 4) * 8;
  const int col   = lane & 15;
  const int kbase = (lane >> 4) * 16;
  const float* Atile = A + (long)mt * 16 * lda;
  v8f acc = {};
  for (int k0 = 0; k0 < K; k0 += 32) {
    v16bf a = frag_a_f32_vec(Atile, lda, row, khalf, k0);
    v16bf b = frag_b_bf16T(WT, K, nt * 16 + col, kbase, k0);
    acc = wmma_bf16(a, b, acc);
  }
  const int n = nt * 16 + (lane & 15);
  const float bv = bias[n];
#pragma unroll
  for (int r = 0; r < 8; ++r) {
    int m = mt * 16 + r + ((lane >= 16) ? 8 : 0);
    float val = acc[r] + bv;
    out[(long)m * ldo + n] = val;
    if (outT) outT[(long)n * M + m] = (__bf16)val;
  }
}

// ---------------- bias = z @ Wb + bb, written as [i][h][j] ----------------
__global__ void bias_z_kernel(const float* __restrict__ z, const __bf16* __restrict__ WbT,
                              const float* __restrict__ bb, float* __restrict__ biasOut) {
  const long mt   = blockIdx.x;   // (N*N)/16 row tiles over flattened (i,j)
  const int lane  = threadIdx.x & 31;
  const int row   = lane & 15;
  const int khalf = (lane >> 4) * 8;
  const int col   = lane & 15;
  const int bcol  = (col < HEADS) ? col : (HEADS - 1);   // clamp; garbage cols store-guarded
  const int kbase = (lane >> 4) * 16;
  const float* Atile = z + mt * 16 * CZ;
  __builtin_prefetch(Atile + (long)(16 + row) * CZ, 0, 1);   // global_prefetch_b8
  v8f acc = {};
  for (int k0 = 0; k0 < CZ; k0 += 32) {
    v16bf a = frag_a_f32_vec(Atile, CZ, row, khalf, k0);
    v16bf b = frag_b_bf16T(WbT, CZ, bcol, kbase, k0);
    acc = wmma_bf16(a, b, acc);
  }
  const int h = lane & 15;
  if (h < HEADS) {
    const float bv = bb[h];
#pragma unroll
    for (int r = 0; r < 8; ++r) {
      long f = mt * 16 + r + ((lane >= 16) ? 8 : 0);
      long i = f / NRES, j = f % NRES;
      biasOut[(i * HEADS + h) * NRES + j] = acc[r] + bv;
    }
  }
}

// ---------------- warp points: tqp/tkp (f32), tvp transposed bf16, sum_p |.|^2 ----------------
__global__ void warp_points_kernel(const float* __restrict__ T,
                                   const float* __restrict__ qp, const float* __restrict__ kp,
                                   const float* __restrict__ vp,
                                   float* __restrict__ tqp, float* __restrict__ tkp,
                                   __bf16* __restrict__ tvpT,
                                   float* __restrict__ sqQ, float* __restrict__ sqK) {
  const int i = blockIdx.x;
  const int t = threadIdx.x;                   // 256 threads
  __shared__ float R[9], tr[3], sq[96];
  if (t < 9) R[t] = T[i * 16 + (t / 3) * 4 + (t % 3)];   // R[x][y]
  if (t < 3) tr[t] = T[i * 16 + t * 4 + 3];
  __syncthreads();
  if (t < 96) {                                 // tqp (0..47) and tkp (48..95)
    const bool isK = (t >= 48);
    const int u = isK ? t - 48 : t;
    const int h = u >> 2, p = u & 3;
    const float* src = isK ? kp : qp;           // (N, 3, H, P): col = x*48 + h*4 + p
    float px = src[i * 144 + 0 * 48 + h * 4 + p];
    float py = src[i * 144 + 1 * 48 + h * 4 + p];
    float pz = src[i * 144 + 2 * 48 + h * 4 + p];
    float wx = R[0] * px + R[1] * py + R[2] * pz + tr[0];
    float wy = R[3] * px + R[4] * py + R[5] * pz + tr[1];
    float wz = R[6] * px + R[7] * py + R[8] * pz + tr[2];
    float* dst = isK ? tkp : tqp;               // [h][i][p*3+x]
    long base = ((long)h * NRES + i) * 12 + p * 3;
    dst[base + 0] = wx; dst[base + 1] = wy; dst[base + 2] = wz;
    sq[t] = wx * wx + wy * wy + wz * wz;
  } else if (t >= 128 && t < 224) {             // tvp, stored transposed bf16: [h*24+pc][j]
    const int u = t - 128;
    const int h = u >> 3, p = u & 7;            // (N, 3, H, PV): col = x*96 + h*8 + p
    float px = vp[i * 288 + 0 * 96 + h * 8 + p];
    float py = vp[i * 288 + 1 * 96 + h * 8 + p];
    float pz = vp[i * 288 + 2 * 96 + h * 8 + p];
    float wx = R[0] * px + R[1] * py + R[2] * pz + tr[0];
    float wy = R[3] * px + R[4] * py + R[5] * pz + tr[1];
    float wz = R[6] * px + R[7] * py + R[8] * pz + tr[2];
    long base = (long)(h * 24 + p * 3) * NRES + i;
    tvpT[base + 0 * NRES] = (__bf16)wx;
    tvpT[base + 1 * NRES] = (__bf16)wy;
    tvpT[base + 2 * NRES] = (__bf16)wz;
  }
  __syncthreads();
  if (t < 12) {
    sqQ[(long)t * NRES + i] = sq[t * 4] + sq[t * 4 + 1] + sq[t * 4 + 2] + sq[t * 4 + 3];
  } else if (t < 24) {
    const int h = t - 12;
    sqK[(long)h * NRES + i] = sq[48 + h * 4] + sq[48 + h * 4 + 1] + sq[48 + h * 4 + 2] + sq[48 + h * 4 + 3];
  }
}

// ---------------- build augmented bf16 operands: Qc/Kc[h][n][32] ----------------
// cols 0..15: q/sqrt(C) (resp. k); 16..27: sqrt(gamma*wc) * warped points; 28..31: zero pad
__global__ void build_comb_kernel(const float* __restrict__ q, const float* __restrict__ k,
                                  const float* __restrict__ tqp, const float* __restrict__ tkp,
                                  const float* __restrict__ hw,
                                  __bf16* __restrict__ Qc, __bf16* __restrict__ Kc) {
  const int h = blockIdx.x, n = blockIdx.y;
  const int t = threadIdx.x;                   // 64
  const float wc = sqrtf(2.0f / (9.0f * NP));
  const float gamma = logf(1.0f + expf(hw[h]));
  const float sg = sqrtf(gamma * wc);
  const bool isK = (t >= 32);
  const int kk = t & 31;
  float val;
  if (kk < 16)      val = isK ? k[(long)n * (HEADS * CH) + h * CH + kk]
                              : q[(long)n * (HEADS * CH) + h * CH + kk] * 0.25f;
  else if (kk < 28) val = (isK ? tkp : tqp)[((long)h * NRES + n) * 12 + (kk - 16)] * sg;
  else              val = 0.0f;
  (isK ? Kc : Qc)[((long)h * NRES + n) * 32 + kk] = (__bf16)val;
}

// ---------------- fused logits: one WMMA per 16x16 tile, in-place over bias ----------------
__global__ void scores_kernel(const __bf16* __restrict__ Qc, const __bf16* __restrict__ Kc,
                              const float* __restrict__ sqQ, const float* __restrict__ sqK,
                              const float* __restrict__ hw, float* __restrict__ logits) {
  const int h = blockIdx.x, it = blockIdx.y;
  const int lane  = threadIdx.x & 31;
  const float wc  = sqrtf(2.0f / (9.0f * NP));
  const float wl  = sqrtf(1.0f / 3.0f);
  const float gamma = logf(1.0f + expf(hw[h]));
  const float hgw = 0.5f * gamma * wc;
  const int row   = lane & 15;
  const int khalf = (lane >> 4) * 8;
  const int col   = lane & 15;
  const int kbase = (lane >> 4) * 16;
  const __bf16* Abase = Qc + ((long)h * NRES + it * 16) * 32;
  const __bf16* Bbase = Kc + (long)h * NRES * 32;
  v16bf afrag = frag_a_bf16_vec(Abase, 32, row, khalf, 0);
  for (int jt = 0; jt < NRES / 16; ++jt) {
    v16bf bfrag = frag_b_bf16T(Bbase, 32, jt * 16 + col, kbase, 0);
    v8f c = {};
    c = wmma_bf16(afrag, bfrag, c);
#pragma unroll
    for (int r = 0; r < 8; ++r) {
      int ii = it * 16 + r + ((lane >= 16) ? 8 : 0);
      int jj = jt * 16 + (lane & 15);
      long idx = ((long)ii * HEADS + h) * NRES + jj;
      logits[idx] = wl * (c[r] + logits[idx]
                          - hgw * (sqQ[(long)h * NRES + ii] + sqK[(long)h * NRES + jj]));
    }
  }
}

// ---------------- softmax over j, emit bf16 att[i][h][j] ----------------
__global__ void softmax_kernel(const float* __restrict__ logits, __bf16* __restrict__ att) {
  const int i = blockIdx.x, h = blockIdx.y;
  const int t = threadIdx.x;                   // 256
  const float* row = logits + ((long)i * HEADS + h) * NRES;
  __shared__ float red[256];
  float m = -1e30f;
  for (int j = t; j < NRES; j += 256) m = fmaxf(m, row[j]);
  red[t] = m; __syncthreads();
  for (int s = 128; s > 0; s >>= 1) { if (t < s) red[t] = fmaxf(red[t], red[t + s]); __syncthreads(); }
  m = red[0]; __syncthreads();
  float sum = 0.0f;
  for (int j = t; j < NRES; j += 256) sum += expf(row[j] - m);
  red[t] = sum; __syncthreads();
  for (int s = 128; s > 0; s >>= 1) { if (t < s) red[t] += red[t + s]; __syncthreads(); }
  const float inv = 1.0f / red[0];
  __bf16* arow = att + ((long)i * HEADS + h) * NRES;
  for (int j = t; j < NRES; j += 256) arow[j] = (__bf16)(expf(row[j] - m) * inv);
}

// ---------------- pairwise_out: per i, att(12x768) @ z_i(768x128) ----------------
// 8 waves per block (one per 16-col tile) share double-buffered z slabs staged into LDS
// with gfx1250 async-to-LDS copies.
__global__ void pairwise_kernel(const __bf16* __restrict__ att, const float* __restrict__ z,
                                float* __restrict__ cat) {
  __shared__ float zbuf[2][32 * CZ];           // 2 x 16KB
  const int i  = blockIdx.x;
  const int t  = threadIdx.x;                  // 256 = 8 waves
  const int ct = t >> 5;                       // wave id -> column tile
  const int lane  = t & 31;
  const int row   = lane & 15;
  const int arow  = (row < HEADS) ? row : (HEADS - 1);
  const int khalf = (lane >> 4) * 8;
  const int col   = lane & 15;
  const int kbase = (lane >> 4) * 16;
  const __bf16* Atile = att + (long)i * HEADS * NRES;
  const float*  zrow  = z + (long)i * NRES * CZ;

  auto stage = [&](int k0, int bufOff) {       // copy 32x128 f32 slab into LDS
#pragma unroll
    for (int r = 0; r < 4; ++r) {
      int flat = r * 1024 + t * 4;
      async_load_b128_to_lds((unsigned)(bufOff + flat * 4), zrow + (long)k0 * CZ + flat);
    }
  };

  stage(0, 0);
  wait_async0();
  __syncthreads();

  v8f acc = {};
  int buf = 0;
  for (int k0 = 0; k0 < NRES; k0 += 32) {
    if (k0 + 32 < NRES) stage(k0 + 32, (buf ^ 1) * (int)sizeof(zbuf[0]));
    v16bf a = frag_a_bf16_vec(Atile, NRES, arow, khalf, k0);
    v16bf b;
#pragma unroll
    for (int e = 0; e < 16; ++e) {
      int kk = kbase + e;
      b[e] = (__bf16)zbuf[buf][kk * CZ + ct * 16 + col];
    }
    acc = wmma_bf16(a, b, acc);
    wait_async0();
    __syncthreads();
    buf ^= 1;
  }
#pragma unroll
  for (int r = 0; r < 8; ++r) {
    int m = r + ((lane >= 16) ? 8 : 0);
    if (m < HEADS)
      cat[(long)i * DCAT + OFF_PAIR + m * CZ + ct * 16 + (lane & 15)] = acc[r];
  }
}

// ---------------- v_out: per (h, i-tile), att(16x768) @ v_h(768x16), B = vT bf16 ----------------
__global__ void attv_kernel(const __bf16* __restrict__ att, const __bf16* __restrict__ vT,
                            float* __restrict__ cat) {
  const int h = blockIdx.x, it = blockIdx.y;
  const int lane  = threadIdx.x & 31;
  const int row   = lane & 15;
  const int khalf = (lane >> 4) * 8;
  const int col   = lane & 15;
  const int kbase = (lane >> 4) * 16;
  const __bf16* Atile = att + ((long)(it * 16) * HEADS + h) * NRES;   // rowStride HEADS*NRES
  const __bf16* Btile = vT + (long)(h * CH) * NRES;                   // vT[n][j]
  v8f acc = {};
  for (int k0 = 0; k0 < NRES; k0 += 32) {
    v16bf a = frag_a_bf16_vec(Atile, (long)HEADS * NRES, row, khalf, k0);
    v16bf b = frag_b_bf16T(Btile, NRES, col, kbase, k0);
    acc = wmma_bf16(a, b, acc);
  }
#pragma unroll
  for (int r = 0; r < 8; ++r) {
    int m = it * 16 + r + ((lane >= 16) ? 8 : 0);
    cat[(long)m * DCAT + OFF_V + h * CH + (lane & 15)] = acc[r];
  }
}

// ---------------- o: per (h, i-tile), att(16x768) @ tvp_h(768x24), B = tvpT bf16 -------------
__global__ void atto_kernel(const __bf16* __restrict__ att, const __bf16* __restrict__ tvpT,
                            float* __restrict__ o_ws) {
  const int h = blockIdx.x, it = blockIdx.y, nt = blockIdx.z;   // nt in {0,1}
  const int lane  = threadIdx.x & 31;
  const int row   = lane & 15;
  const int khalf = (lane >> 4) * 8;
  const int colAbs = nt * 16 + (lane & 15);
  const int bcol  = (colAbs < 24) ? colAbs : 23;                // clamp pad cols
  const int kbase = (lane >> 4) * 16;
  const __bf16* Atile = att + ((long)(it * 16) * HEADS + h) * NRES;
  const __bf16* Btile = tvpT + (long)h * 24 * NRES;             // tvpT[h*24+pc][j]
  v8f acc = {};
  for (int k0 = 0; k0 < NRES; k0 += 32) {
    v16bf a = frag_a_bf16_vec(Atile, (long)HEADS * NRES, row, khalf, k0);
    v16bf b = frag_b_bf16T(Btile, NRES, bcol, kbase, k0);
    acc = wmma_bf16(a, b, acc);
  }
#pragma unroll
  for (int r = 0; r < 8; ++r) {
    int m  = it * 16 + r + ((lane >= 16) ? 8 : 0);
    int nn = nt * 16 + (lane & 15);
    if (nn < 24) o_ws[((long)h * NRES + m) * 24 + nn] = acc[r];
  }
}

// ---------------- inverse warp of o, write vp_out + vp_norm into cat ----------------
__global__ void finalize_vp_kernel(const float* __restrict__ T, const float* __restrict__ o_ws,
                                   float* __restrict__ cat) {
  const int i = blockIdx.x;
  const int t = threadIdx.x;                   // 128 (96 active)
  __shared__ float R[9], tr[3];
  if (t < 9) R[t] = T[i * 16 + (t / 3) * 4 + (t % 3)];
  if (t < 3) tr[t] = T[i * 16 + t * 4 + 3];
  __syncthreads();
  if (t < 96) {
    const int h = t >> 3, p = t & 7;
    long base = ((long)h * NRES + i) * 24 + p * 3;
    float ox = o_ws[base + 0] - tr[0];
    float oy = o_ws[base + 1] - tr[1];
    float oz = o_ws[base + 2] - tr[2];
    float ux = R[0] * ox + R[3] * oy + R[6] * oz;   // out_x = sum_y R[y][x]*o_y
    float uy = R[1] * ox + R[4] * oy + R[7] * oz;
    float uz = R[2] * ox + R[5] * oy + R[8] * oz;
    long cb = (long)i * DCAT;
    cat[cb + OFF_VP + 0 * 96 + h * 8 + p] = ux;
    cat[cb + OFF_VP + 1 * 96 + h * 8 + p] = uy;
    cat[cb + OFF_VP + 2 * 96 + h * 8 + p] = uz;
    cat[cb + OFF_NORM + h * 8 + p] = sqrtf(ux * ux + uy * uy + uz * uz);
  }
}

// ---------------- host launch ----------------
extern "C" void kernel_launch(void* const* d_in, const int* in_sizes, int n_in,
                              void* d_out, int out_size, void* d_ws, size_t ws_size,
                              hipStream_t stream) {
  const float* s    = (const float*)d_in[0];
  const float* z    = (const float*)d_in[1];
  const float* T    = (const float*)d_in[2];
  const float* Wq   = (const float*)d_in[3];  const float* bq   = (const float*)d_in[4];
  const float* Wk   = (const float*)d_in[5];  const float* bk   = (const float*)d_in[6];
  const float* Wv   = (const float*)d_in[7];  const float* bv   = (const float*)d_in[8];
  const float* Wqp  = (const float*)d_in[9];  const float* bqp  = (const float*)d_in[10];
  const float* Wkp  = (const float*)d_in[11]; const float* bkp  = (const float*)d_in[12];
  const float* Wvp  = (const float*)d_in[13]; const float* bvp  = (const float*)d_in[14];
  const float* Wb   = (const float*)d_in[15]; const float* bb   = (const float*)d_in[16];
  const float* Wout = (const float*)d_in[17]; const float* bout = (const float*)d_in[18];
  const float* hw   = (const float*)d_in[19];
  float* out = (float*)d_out;

  char* wptr = (char*)d_ws;
  auto alloc = [&](size_t bytes) -> void* {
    void* p = (void*)wptr;
    wptr += (bytes + 255) & ~(size_t)255;
    return p;
  };
  // f32 buffers
  float* qws    = (float*)alloc((size_t)NRES * 192 * 4);
  float* kws    = (float*)alloc((size_t)NRES * 192 * 4);
  float* vws    = (float*)alloc((size_t)NRES * 192 * 4);
  float* qpws   = (float*)alloc((size_t)NRES * 144 * 4);
  float* kpws   = (float*)alloc((size_t)NRES * 144 * 4);
  float* vpws   = (float*)alloc((size_t)NRES * 288 * 4);
  float* tqpws  = (float*)alloc((size_t)HEADS * NRES * 12 * 4);
  float* tkpws  = (float*)alloc((size_t)HEADS * NRES * 12 * 4);
  float* sqQws  = (float*)alloc((size_t)HEADS * NRES * 4);
  float* sqKws  = (float*)alloc((size_t)HEADS * NRES * 4);
  float* logits = (float*)alloc((size_t)NRES * HEADS * NRES * 4);   // bias, then in-place logits
  float* ows    = (float*)alloc((size_t)HEADS * NRES * 24 * 4);
  float* cat    = (float*)alloc((size_t)NRES * DCAT * 4);
  // bf16 buffers
  __bf16* att   = (__bf16*)alloc((size_t)NRES * HEADS * NRES * 2);
  __bf16* WqT   = (__bf16*)alloc((size_t)CS * 192 * 2);
  __bf16* WkT   = (__bf16*)alloc((size_t)CS * 192 * 2);
  __bf16* WvT   = (__bf16*)alloc((size_t)CS * 192 * 2);
  __bf16* WqpT  = (__bf16*)alloc((size_t)CS * 144 * 2);
  __bf16* WkpT  = (__bf16*)alloc((size_t)CS * 144 * 2);
  __bf16* WvpT  = (__bf16*)alloc((size_t)CS * 288 * 2);
  __bf16* WbT   = (__bf16*)alloc((size_t)CZ * HEADS * 2);
  __bf16* WoutT = (__bf16*)alloc((size_t)DCAT * CS * 2);
  __bf16* vT    = (__bf16*)alloc((size_t)192 * NRES * 2);
  __bf16* tvpT  = (__bf16*)alloc((size_t)HEADS * 24 * NRES * 2);
  __bf16* Qc    = (__bf16*)alloc((size_t)HEADS * NRES * 32 * 2);
  __bf16* Kc    = (__bf16*)alloc((size_t)HEADS * NRES * 32 * 2);

  const dim3 wave(32);
  auto tgrid = [](long n) { return dim3((unsigned)((n + 255) / 256)); };

  // 0) one-off bf16 weight transposes
  transpose_bf16_kernel<<<tgrid((long)CS * 192), 256, 0, stream>>>(Wq,  CS, 192, WqT);
  transpose_bf16_kernel<<<tgrid((long)CS * 192), 256, 0, stream>>>(Wk,  CS, 192, WkT);
  transpose_bf16_kernel<<<tgrid((long)CS * 192), 256, 0, stream>>>(Wv,  CS, 192, WvT);
  transpose_bf16_kernel<<<tgrid((long)CS * 144), 256, 0, stream>>>(Wqp, CS, 144, WqpT);
  transpose_bf16_kernel<<<tgrid((long)CS * 144), 256, 0, stream>>>(Wkp, CS, 144, WkpT);
  transpose_bf16_kernel<<<tgrid((long)CS * 288), 256, 0, stream>>>(Wvp, CS, 288, WvpT);
  transpose_bf16_kernel<<<tgrid((long)CZ * HEADS), 256, 0, stream>>>(Wb, CZ, HEADS, WbT);
  transpose_bf16_kernel<<<tgrid((long)DCAT * CS), 256, 0, stream>>>(Wout, DCAT, CS, WoutT);

  // 1) projections (WMMA; v also emitted transposed bf16)
  gemm_bias_kernel<<<dim3(NRES / 16, 192 / 16), wave, 0, stream>>>(s, CS, WqT,  bq,  qws,  192, CS, nullptr, NRES);
  gemm_bias_kernel<<<dim3(NRES / 16, 192 / 16), wave, 0, stream>>>(s, CS, WkT,  bk,  kws,  192, CS, nullptr, NRES);
  gemm_bias_kernel<<<dim3(NRES / 16, 192 / 16), wave, 0, stream>>>(s, CS, WvT,  bv,  vws,  192, CS, vT, NRES);
  gemm_bias_kernel<<<dim3(NRES / 16, 144 / 16), wave, 0, stream>>>(s, CS, WqpT, bqp, qpws, 144, CS, nullptr, NRES);
  gemm_bias_kernel<<<dim3(NRES / 16, 144 / 16), wave, 0, stream>>>(s, CS, WkpT, bkp, kpws, 144, CS, nullptr, NRES);
  gemm_bias_kernel<<<dim3(NRES / 16, 288 / 16), wave, 0, stream>>>(s, CS, WvpT, bvp, vpws, 288, CS, nullptr, NRES);

  // 2) bias = z @ Wb + bb  (dominant z read #1, vectorized b128 stream)
  bias_z_kernel<<<dim3((NRES * NRES) / 16), wave, 0, stream>>>(z, WbT, bb, logits);

  // 3) point warps + squared norms (+ transposed bf16 tvp)
  warp_points_kernel<<<dim3(NRES), dim3(256), 0, stream>>>(T, qpws, kpws, vpws,
                                                           tqpws, tkpws, tvpT, sqQws, sqKws);

  // 4) augmented operands + fused logits (qk + point cross-term in one WMMA)
  build_comb_kernel<<<dim3(HEADS, NRES), dim3(64), 0, stream>>>(qws, kws, tqpws, tkpws, hw, Qc, Kc);
  scores_kernel<<<dim3(HEADS, NRES / 16), wave, 0, stream>>>(Qc, Kc, sqQws, sqKws, hw, logits);

  // 5) softmax -> bf16 attention
  softmax_kernel<<<dim3(NRES, HEADS), dim3(256), 0, stream>>>(logits, att);

  // 6) attention-weighted outputs (pairwise = dominant z read #2, async-LDS staged)
  pairwise_kernel<<<dim3(NRES), dim3(256), 0, stream>>>(att, z, cat);
  attv_kernel<<<dim3(HEADS, NRES / 16), wave, 0, stream>>>(att, vT, cat);
  atto_kernel<<<dim3(HEADS, NRES / 16, 2), wave, 0, stream>>>(att, tvpT, ows);

  // 7) inverse warp + norms into cat
  finalize_vp_kernel<<<dim3(NRES), dim3(128), 0, stream>>>(T, ows, cat);

  // 8) final projection: out = cat @ Wout + bout
  gemm_bias_kernel<<<dim3(NRES / 16, CS / 16), wave, 0, stream>>>(cat, DCAT, WoutT, bout,
                                                                  out, CS, DCAT, nullptr, NRES);
}
